// NoisyTopkRouter_53841710022745
// MI455X (gfx1250) — matrix operand order repaired and checked by reading, not compiled, not run on hardware
//
#include <hip/hip_runtime.h>
#include <hip/hip_bf16.h>
#include <math.h>

// ---------------------------------------------------------------------------
// NoisyTopkRouter, fused for MI455X (gfx1250, wave32, WMMA bf16 16x16x32)
//   logits = h @ Wg^T ; scale = softplus(h @ Wn^T) ; logits += noise*scale
//   gates = softmax(logits) ; vals,inds = top8(gates)
// ---------------------------------------------------------------------------

typedef __attribute__((ext_vector_type(16))) __bf16  v16bf;
typedef __attribute__((ext_vector_type(8)))  float   v8f;
typedef __attribute__((ext_vector_type(4)))  float   v4f;

#define D_DIM        2048
#define N_EXPERTS    64
#define TOPK         8
#define WAVES_PER_WG 4
#define TOK_PER_WAVE 16
#define TOK_PER_WG   (WAVES_PER_WG * TOK_PER_WAVE)   // 64 tokens / workgroup
#define LDS_STRIDE   68                              // 64 + pad (dwords) to break bank conflicts
#define TOTAL_TOKENS (4 * 4096)                      // B*S

// raw 32-bit LDS byte offset of a __shared__ pointer (for async-to-LDS asm)
__device__ __forceinline__ unsigned lds_off(void* p) {
    return (unsigned)(unsigned long long)(__attribute__((address_space(3))) void*)p;
}

// ---------------------------------------------------------------------------
// Kernel 0: one-time fp32 -> bf16 weight repack into workspace.
// Layout: wB[0 .. 64*2048)      = W_gate  (expert-major, k contiguous)
//         wB[64*2048 .. 2*...)  = W_noise
// ---------------------------------------------------------------------------
__global__ void prep_weights_bf16(const float* __restrict__ Wg,
                                  const float* __restrict__ Wn,
                                  __bf16* __restrict__ wB) {
    const int PER = N_EXPERTS * D_DIM;                  // 131072
    int i = blockIdx.x * blockDim.x + threadIdx.x;
    if (i < PER) {
        wB[i]       = (__bf16)Wg[i];
        wB[PER + i] = (__bf16)Wn[i];
    }
}

// ---------------------------------------------------------------------------
// Kernel 1: fused GEMM (bf16 WMMA) + noisy-softplus + softmax + top-8
// One workgroup = 4 waves, 16 tokens per wave, full 64 experts per wave.
// ---------------------------------------------------------------------------
__global__ __launch_bounds__(WAVES_PER_WG * 32)
void router_kernel(const float*  __restrict__ h,       // [B*S, D] fp32
                   const __bf16* __restrict__ wB,      // packed weights (gate,noise)
                   const float*  __restrict__ noise,   // [B*S, 64] fp32
                   float* __restrict__ out_vals,       // [B*S, 8]
                   int*   __restrict__ out_inds,       // [B*S, 8]
                   float* __restrict__ out_gates)      // [B*S, 64]
{
    __shared__ float s_noise[TOK_PER_WG * LDS_STRIDE];
    __shared__ float s_logit[TOK_PER_WG * LDS_STRIDE];

    const int tid   = threadIdx.x;
    const int lane  = tid & 31;
    const int wave  = tid >> 5;
    const int wgTok = blockIdx.x * TOK_PER_WG;

    // ---- stage this WG's noise tile into LDS asynchronously (overlaps GEMM) ----
    {
        const float* nb    = noise + (size_t)wgTok * N_EXPERTS;
        unsigned     sbase = lds_off((void*)s_noise);
        // 64 rows * 256 B = 1024 x 16B chunks, 128 threads -> 8 chunks each
        #pragma unroll
        for (int i = 0; i < 8; ++i) {
            int c = tid + i * (WAVES_PER_WG * 32);
            unsigned ldsA = sbase + (unsigned)((c >> 4) * (LDS_STRIDE * 4) + (c & 15) * 16);
            int voff = c * 16;   // global rows are contiguous (64 floats)
            asm volatile("global_load_async_to_lds_b128 %0, %1, %2"
                         :: "v"(ldsA), "v"(voff), "s"(nb) : "memory");
        }
    }

    // ---- bf16 WMMA GEMM: D = A(16x32 fp32->bf16) x B(32x16 bf16) + C ----
    // A as loaded here (16x32, contiguous k, same order as B): lanes 0-15: K 0-15 ; lanes 16-31: K 16-31
    const int    rowA = wgTok + wave * TOK_PER_WAVE + (lane & 15);
    const float* aRow = h + (size_t)rowA * D_DIM + ((lane >> 4) * 16);
    // B as loaded here (32x16, contiguous k, same order as A): lanes 0-15 hold K 0-15, lanes 16-31 hold K 16-31
    const int eLane = lane & 15;
    const int kBsel = (lane >> 4) * 16;

    v8f acc[8] = {};   // tiles 0-3: gate experts [0,16,32,48]+ ; tiles 4-7: noise

    #pragma unroll 2
    for (int kk = 0; kk < D_DIM; kk += 32) {
        v4f a0 = *(const v4f*)(aRow + kk);
        v4f a1 = *(const v4f*)(aRow + kk + 4);
        v4f a2 = *(const v4f*)(aRow + kk + 8);
        v4f a3 = *(const v4f*)(aRow + kk + 12);
        v16bf a;
        #pragma unroll
        for (int i = 0; i < 4; ++i) {
            a[i]      = (__bf16)a0[i];
            a[4 + i]  = (__bf16)a1[i];
            a[8 + i]  = (__bf16)a2[i];
            a[12 + i] = (__bf16)a3[i];
        }
        #pragma unroll
        for (int t = 0; t < 8; ++t) {
            const int e = (t & 3) * 16 + eLane;                       // expert column
            const __bf16* wp = wB + (size_t)(t >> 2) * (N_EXPERTS * D_DIM)
                                  + (size_t)e * D_DIM + kk + kBsel;
            v16bf b = *(const v16bf*)wp;                              // 16 sequential K
            acc[t] = __builtin_amdgcn_wmma_f32_16x16x32_bf16(
                         false, a, false, b, (short)0, acc[t], false, false);
        }
    }

    // ---- make staged noise visible, then fused noisy epilogue ----
    asm volatile("s_wait_asynccnt 0x0" ::: "memory");
    __syncthreads();

    // C/D layout: lanes 0-15 -> N=lane, M=r ; lanes 16-31 -> N=lane-16, M=8+r
    const int mHalf = (lane >> 4) * 8;
    #pragma unroll
    for (int t = 0; t < 4; ++t) {
        const int n = t * 16 + eLane;
        #pragma unroll
        for (int r = 0; r < 8; ++r) {
            const int   mt = wave * TOK_PER_WAVE + mHalf + r;   // local token
            const float gl = acc[t][r];
            const float nl = acc[t + 4][r];
            const float sp = (nl > 20.f) ? nl : log1pf(__expf(nl));   // softplus
            s_logit[mt * LDS_STRIDE + n] = gl + s_noise[mt * LDS_STRIDE + n] * sp;
        }
    }
    __syncthreads();

    // ---- per-token softmax + top-8 (one thread per token) ----
    if (tid < TOK_PER_WG) {
        float*       row  = s_logit + tid * LDS_STRIDE;
        const size_t gtok = (size_t)wgTok + tid;

        float mx = -3.402823466e38f;
        #pragma unroll 8
        for (int e = 0; e < N_EXPERTS; ++e) mx = fmaxf(mx, row[e]);

        float sum = 0.f;
        #pragma unroll 8
        for (int e = 0; e < N_EXPERTS; ++e) {
            float g = __expf(row[e] - mx);
            sum += g;
            row[e] = g;
        }
        const float inv  = 1.f / sum;
        float*      gout = out_gates + gtok * N_EXPERTS;
        #pragma unroll 8
        for (int e = 0; e < N_EXPERTS; ++e) {
            float g = row[e] * inv;
            row[e]  = g;
            gout[e] = g;
        }

        // top-8 by repeated arg-max (gates in (0,1], sentinel -1 is safe)
        float* vout = out_vals + gtok * TOPK;
        int*   iout = out_inds + gtok * TOPK;
        #pragma unroll
        for (int k = 0; k < TOPK; ++k) {
            float best = -1.f; int bi = 0;
            #pragma unroll 8
            for (int e = 0; e < N_EXPERTS; ++e) {
                float v = row[e];
                if (v > best) { best = v; bi = e; }
            }
            row[bi] = -1.f;
            vout[k] = best;
            iout[k] = bi;
        }
    }
}

// ---------------------------------------------------------------------------
extern "C" void kernel_launch(void* const* d_in, const int* in_sizes, int n_in,
                              void* d_out, int out_size, void* d_ws, size_t ws_size,
                              hipStream_t stream) {
    (void)in_sizes; (void)n_in; (void)out_size; (void)ws_size;

    const float* h  = (const float*)d_in[0];   // hidden_states [4,4096,2048]
    const float* Wg = (const float*)d_in[1];   // W_gate  [64,2048]
    const float* Wn = (const float*)d_in[2];   // W_noise [64,2048]
    const float* nz = (const float*)d_in[3];   // noise   [4,4096,64]

    float* out   = (float*)d_out;              // vals ++ inds ++ gates
    float* vals  = out;
    int*   inds  = (int*)(out + (size_t)TOTAL_TOKENS * TOPK);
    float* gates = out + (size_t)TOTAL_TOKENS * TOPK * 2;

    __bf16* wB = (__bf16*)d_ws;                // 2*64*2048 bf16 = 512 KB scratch

    hipLaunchKernelGGL(prep_weights_bf16,
                       dim3((N_EXPERTS * D_DIM + 255) / 256), dim3(256), 0, stream,
                       Wg, Wn, wB);

    hipLaunchKernelGGL(router_kernel,
                       dim3(TOTAL_TOKENS / TOK_PER_WG), dim3(WAVES_PER_WG * 32), 0, stream,
                       h, wB, nz, vals, inds, gates);
}
